// QWindowAttention_27273042330109
// MI455X (gfx1250) — matrix operand order repaired and checked by reading, not compiled, not run on hardware
//
#include <hip/hip_runtime.h>
#include <hip/hip_bf16.h>

typedef _Float16 h8  __attribute__((ext_vector_type(8)));
typedef _Float16 h16 __attribute__((ext_vector_type(16)));
typedef float    f8  __attribute__((ext_vector_type(8)));

#define NTOK 49
#define NH   16
#define HD   32
#define CDIM 512
#define NWIN 4096
#define QK_SCALE 0.17677669529663687f /* 32^-0.5 */

static __device__ __forceinline__ h16 cat8(h8 lo, h8 hi) {
    return __builtin_shufflevector(lo, hi, 0,1,2,3,4,5,6,7,8,9,10,11,12,13,14,15);
}

// A fragment (16x32, f16). ISA layout: lanes 0-15 hold K=[0..7],[16..23];
// lanes 16-31 hold K=[8..15],[24..31]. 'row' = M row for this lane (l16).
static __device__ __forceinline__ h16 fragA(const _Float16* base, int row, int stride,
                                            int k0, int hf) {
    const _Float16* p = base + row * stride + k0 + hf * 8;
    h8 lo = *(const h8*)(p);
    h8 hi = *(const h8*)(p + 16);
    return cat8(lo, hi);
}

// B fragment (32x16, f16). ISA layout: lanes 0-15 hold K=[0..7],[16..23];
// lanes 16-31 hold K=[8..15],[24..31]. 'col' = N column; source row 'col' holds K contiguously.
static __device__ __forceinline__ h16 fragB(const _Float16* base, int col, int stride,
                                            int k0, int hf) {
    const _Float16* p = base + col * stride + k0 + hf * 8;
    h8 lo = *(const h8*)(p);
    h8 hi = *(const h8*)(p + 16);
    return cat8(lo, hi);
}

static __device__ __forceinline__ f8 wmma_f16(h16 a, h16 b, f8 c) {
    return __builtin_amdgcn_wmma_f32_16x16x32_f16(false, a, false, b, (short)0, c,
                                                  false, false);
}

// Async copy: 16B global -> LDS, tracked on ASYNCcnt (CDNA5 path).
// GV addressing: VDST = LDS byte offset (low 32 bits of flat shared address),
// VADDR = 64-bit global address pair, SADDR = off (null).
static __device__ __forceinline__ void async_cp16(_Float16* lds_dst,
                                                  const _Float16* gsrc) {
    unsigned lds_off = (unsigned)(uintptr_t)lds_dst;
    asm volatile("global_load_async_to_lds_b128 %0, %1, off"
                 :: "v"(lds_off), "v"(gsrc)
                 : "memory");
}
static __device__ __forceinline__ void async_wait0() {
    asm volatile("s_wait_asynccnt 0" ::: "memory");
}

// ---------------------------------------------------------------------------
// Kernel 0: weight f32->f16 conversion + relative-position bias table expand.
// ---------------------------------------------------------------------------
__global__ __launch_bounds__(256) void prep_kernel(const float* __restrict__ qkv_w,
                                                   const float* __restrict__ proj_w,
                                                   const float* __restrict__ rpb,
                                                   _Float16* __restrict__ wq,
                                                   _Float16* __restrict__ wp,
                                                   float* __restrict__ biasf) {
    int i = blockIdx.x * 256 + threadIdx.x;
    if (i < 3 * CDIM * CDIM) { wq[i] = (_Float16)qkv_w[i]; return; }
    i -= 3 * CDIM * CDIM;
    if (i < CDIM * CDIM) { wp[i] = (_Float16)proj_w[i]; return; }
    i -= CDIM * CDIM;
    if (i < NH * NTOK * NTOK) {
        int h = i / (NTOK * NTOK);
        int r = i - h * (NTOK * NTOK);
        int m = r / NTOK, n = r - (r / NTOK) * NTOK;
        int mh = m / 7, mw = m - mh * 7;
        int nh = n / 7, nw = n - nh * 7;
        int idx = (mh - nh + 6) * 13 + (mw - nw + 6);
        biasf[i] = rpb[idx * NH + h];
    }
}

// ---------------------------------------------------------------------------
// Kernel 1: QKV projection GEMM.  M=200704, N=1536, K=512.
// 128x128 block tile, BK=32 (one WMMA K-step), 8 waves = 4(m) x 2(n).
// Register-double-buffered: next K tile's global loads are issued before the
// WMMAs and only consumed (waited) at the top of the next iteration.
// Epilogue routes to q[b][h][t][d]*scale, k[b][h][t][d], v^T[b][h][d][t] (f16).
// ---------------------------------------------------------------------------
__global__ __launch_bounds__(256) void qkv_gemm_kernel(const float* __restrict__ x,
                                                       const _Float16* __restrict__ wq,
                                                       const float* __restrict__ qb,
                                                       _Float16* __restrict__ q16,
                                                       _Float16* __restrict__ k16,
                                                       _Float16* __restrict__ vt16) {
    __shared__ __align__(16) _Float16 As[128 * 40];
    __shared__ __align__(16) _Float16 Bs[128 * 40];
    const int tid  = threadIdx.x;
    const int lane = tid & 31, wave = tid >> 5;
    const int l16  = lane & 15, hf = lane >> 4;
    const int wm = wave & 3;   // 4 wave-rows of 32
    const int wn = wave >> 2;  // 2 wave-cols of 64
    const int gm0 = blockIdx.y * 128;
    const int gn0 = blockIdx.x * 128;

    const f8 zf = {0.f,0.f,0.f,0.f,0.f,0.f,0.f,0.f};
    f8 acc[2][4];
#pragma unroll
    for (int mi = 0; mi < 2; ++mi)
#pragma unroll
        for (int nj = 0; nj < 4; ++nj) acc[mi][nj] = zf;

    float4 va[4];
    h8     vb[2];
    // Prologue: load K tile 0 into registers.
#pragma unroll
    for (int i = 0; i < 4; ++i) {
        int e = tid + i * 256, row = e >> 3, c = e & 7;
        va[i] = *(const float4*)&x[(size_t)(gm0 + row) * CDIM + c * 4];
    }
#pragma unroll
    for (int i = 0; i < 2; ++i) {
        int e = tid + i * 256, row = e >> 2, c = e & 3;
        vb[i] = *(const h8*)&wq[(size_t)(gn0 + row) * CDIM + c * 8];
    }

    for (int k0 = 0; k0 < CDIM; k0 += 32) {
        // Commit staged registers to LDS (f32->f16 convert for A).
#pragma unroll
        for (int i = 0; i < 4; ++i) {
            int e = tid + i * 256, row = e >> 3, c = e & 7;
            _Float16* d = &As[row * 40 + c * 4];
            d[0] = (_Float16)va[i].x; d[1] = (_Float16)va[i].y;
            d[2] = (_Float16)va[i].z; d[3] = (_Float16)va[i].w;
        }
#pragma unroll
        for (int i = 0; i < 2; ++i) {
            int e = tid + i * 256, row = e >> 2, c = e & 3;
            *(h8*)&Bs[row * 40 + c * 8] = vb[i];
        }
        __syncthreads();

        // Issue next tile's loads now; they complete while WMMAs run.
        if (k0 + 32 < CDIM) {
#pragma unroll
            for (int i = 0; i < 4; ++i) {
                int e = tid + i * 256, row = e >> 3, c = e & 7;
                va[i] = *(const float4*)&x[(size_t)(gm0 + row) * CDIM +
                                           k0 + 32 + c * 4];
            }
#pragma unroll
            for (int i = 0; i < 2; ++i) {
                int e = tid + i * 256, row = e >> 2, c = e & 3;
                vb[i] = *(const h8*)&wq[(size_t)(gn0 + row) * CDIM +
                                        k0 + 32 + c * 8];
            }
            if (k0 + 64 < CDIM) {   // near-scope prefetch two tiles ahead
                int prow = tid >> 1, pc = tid & 1;
                __builtin_prefetch(&x[(size_t)(gm0 + prow) * CDIM +
                                      k0 + 64 + pc * 16], 0, 3);
            }
        }

        h16 af[2], bf[4];
#pragma unroll
        for (int mi = 0; mi < 2; ++mi)
            af[mi] = fragA(As, wm * 32 + mi * 16 + l16, 40, 0, hf);
#pragma unroll
        for (int nj = 0; nj < 4; ++nj)
            bf[nj] = fragB(Bs, wn * 64 + nj * 16 + l16, 40, 0, hf);
#pragma unroll
        for (int mi = 0; mi < 2; ++mi)
#pragma unroll
            for (int nj = 0; nj < 4; ++nj)
                acc[mi][nj] = wmma_f16(af[mi], bf[nj], acc[mi][nj]);
        __syncthreads();
    }

    // Epilogue: bias add + scatter to q/k/v^T f16 buffers
    int   ncol[4];
    float bn[4];
#pragma unroll
    for (int nj = 0; nj < 4; ++nj) {
        ncol[nj] = gn0 + wn * 64 + nj * 16 + l16;
        bn[nj]   = qb[ncol[nj]];
    }
#pragma unroll
    for (int mi = 0; mi < 2; ++mi) {
#pragma unroll
        for (int r = 0; r < 8; ++r) {
            int m = gm0 + wm * 32 + mi * 16 + r + hf * 8;
            int b = m / NTOK, t = m - b * NTOK;
#pragma unroll
            for (int nj = 0; nj < 4; ++nj) {
                int n = ncol[nj];
                float v = acc[mi][nj][r] + bn[nj];
                int sel = n >> 9;         // 0=q 1=k 2=v
                int nn  = n & 511;
                int h   = nn >> 5, d = nn & 31;
                size_t bh = ((size_t)b * NH + h) * (NTOK * HD);
                if (sel == 0)      q16[bh + t * HD + d] = (_Float16)(v * QK_SCALE);
                else if (sel == 1) k16[bh + t * HD + d] = (_Float16)v;
                else               vt16[bh + d * NTOK + t] = (_Float16)v;
            }
        }
    }
}

// ---------------------------------------------------------------------------
// Kernel 2: per-window attention.  One block per window, one wave per 2 heads.
// attn = (q*scale) k^T + bias + mask ; softmax ; out = attn v.
// ---------------------------------------------------------------------------
__global__ __launch_bounds__(256) void attn_kernel(const _Float16* __restrict__ q16,
                                                   const _Float16* __restrict__ k16,
                                                   const _Float16* __restrict__ vt16,
                                                   const float* __restrict__ mask,
                                                   const float* __restrict__ biasf,
                                                   _Float16* __restrict__ y16) {
    __shared__ __align__(16) _Float16 qs[8][64 * 32];
    __shared__ __align__(16) _Float16 ks[8][64 * 32];
    __shared__ __align__(16) _Float16 vs[8][32 * 64];
    __shared__ __align__(16) _Float16 as_[8][16 * 64];

    const int b    = blockIdx.x;
    const int tid  = threadIdx.x;
    const int lane = tid & 31, wave = tid >> 5;
    const int l16  = lane & 15, hf = lane >> 4;
    const int wi   = b & 63;  // shift-window mask index (nW = 64)
    const float* maskw = mask + (size_t)wi * NTOK * NTOK;

    _Float16* qsl = &qs[wave][0];
    _Float16* ksl = &ks[wave][0];
    _Float16* vsl = &vs[wave][0];
    _Float16* asl = &as_[wave][0];

    const h8 hz = {(_Float16)0,(_Float16)0,(_Float16)0,(_Float16)0,
                   (_Float16)0,(_Float16)0,(_Float16)0,(_Float16)0};
    const f8 zf = {0.f,0.f,0.f,0.f,0.f,0.f,0.f,0.f};

    for (int hi = 0; hi < 2; ++hi) {
        const int h = wave * 2 + hi;
        const size_t bh = ((size_t)b * NH + h) * (NTOK * HD);
        const _Float16* qg = q16 + bh;
        const _Float16* kg = k16 + bh;
        const _Float16* vg = vt16 + bh;

        // ---- stage q/k rows (zero-pad tokens 49..63) ----
#pragma unroll
        for (int rr = 0; rr < 2; ++rr) {
            int t = lane * 2 + rr;
            if (t < NTOK) {
#pragma unroll
                for (int c = 0; c < 4; ++c) {
                    *(h8*)&qsl[t * 32 + c * 8] = *(const h8*)&qg[t * HD + c * 8];
                    *(h8*)&ksl[t * 32 + c * 8] = *(const h8*)&kg[t * HD + c * 8];
                }
            } else {
#pragma unroll
                for (int c = 0; c < 4; ++c) {
                    *(h8*)&qsl[t * 32 + c * 8] = hz;
                    *(h8*)&ksl[t * 32 + c * 8] = hz;
                }
            }
        }
        // ---- stage v^T rows (d = lane), zero-pad token columns ----
        {
            int d = lane;
            for (int t = 0; t < 64; ++t)
                vsl[d * 64 + t] = (t < NTOK) ? vg[d * NTOK + t] : (_Float16)0;
        }

        // ---- 4 stripes of 16 rows ----
        for (int ms = 0; ms < 4; ++ms) {
            // gemm1: (16x32) x (32x64) in 4 WMMAs, K=32 each
            f8 cc[4] = {zf, zf, zf, zf};
            h16 aq = fragA(qsl, ms * 16 + l16, 32, 0, hf);
#pragma unroll
            for (int j = 0; j < 4; ++j) {
                h16 bk = fragB(ksl, j * 16 + l16, 32, 0, hf);
                cc[j] = wmma_f16(aq, bk, cc[j]);
            }

            // bias + mask + padding, then softmax along the row (16 lanes/half)
            float e_[4][8];
#pragma unroll
            for (int j = 0; j < 4; ++j) {
                int n = j * 16 + l16;
#pragma unroll
                for (int r = 0; r < 8; ++r) {
                    int mg = ms * 16 + r + hf * 8;
                    float v = cc[j][r];
                    if (mg < NTOK && n < NTOK)
                        v += biasf[h * (NTOK * NTOK) + mg * NTOK + n] +
                             maskw[mg * NTOK + n];
                    else
                        v = -1e30f;
                    e_[j][r] = v;
                }
            }
            float rm[8], rs[8];
#pragma unroll
            for (int r = 0; r < 8; ++r) {
                float m0 = fmaxf(fmaxf(e_[0][r], e_[1][r]), fmaxf(e_[2][r], e_[3][r]));
#pragma unroll
                for (int s = 1; s < 16; s <<= 1)
                    m0 = fmaxf(m0, __shfl_xor(m0, s, 32));
                rm[r] = m0;
            }
#pragma unroll
            for (int j = 0; j < 4; ++j)
#pragma unroll
                for (int r = 0; r < 8; ++r)
                    e_[j][r] = __expf(e_[j][r] - rm[r]);
#pragma unroll
            for (int r = 0; r < 8; ++r) {
                float s0 = (e_[0][r] + e_[1][r]) + (e_[2][r] + e_[3][r]);
#pragma unroll
                for (int s = 1; s < 16; s <<= 1)
                    s0 += __shfl_xor(s0, s, 32);
                rs[r] = __builtin_amdgcn_rcpf(s0);
            }
            // write probability stripe (f16) to LDS for gemm2's A fragments
#pragma unroll
            for (int j = 0; j < 4; ++j)
#pragma unroll
                for (int r = 0; r < 8; ++r)
                    asl[(r + hf * 8) * 64 + j * 16 + l16] =
                        (_Float16)(e_[j][r] * rs[r]);

            // gemm2: (16x64) x (64x32) -> 4 WMMAs (2 K-steps x 2 N-frags)
            f8 o[2] = {zf, zf};
#pragma unroll
            for (int kh = 0; kh < 2; ++kh) {
                h16 ap = fragA(asl, l16, 64, kh * 32, hf);
#pragma unroll
                for (int jj = 0; jj < 2; ++jj) {
                    h16 bv = fragB(vsl, jj * 16 + l16, 64, kh * 32, hf);
                    o[jj] = wmma_f16(ap, bv, o[jj]);
                }
            }
            // scatter valid rows to attention output (token-major, f16)
#pragma unroll
            for (int jj = 0; jj < 2; ++jj) {
                int d = jj * 16 + l16;
#pragma unroll
                for (int r = 0; r < 8; ++r) {
                    int mg = ms * 16 + r + hf * 8;
                    if (mg < NTOK)
                        y16[((size_t)(b * NTOK + mg)) * CDIM + h * HD + d] =
                            (_Float16)o[jj][r];
                }
            }
        }
    }
}

// ---------------------------------------------------------------------------
// Kernel 3: output projection GEMM. M=200704, N=512, K=512, A is f16 already.
// Double-buffered async DMA staging: GLOBAL_LOAD_ASYNC_TO_LDS_B128 for tile
// i+1 is issued into the alternate LDS buffer before computing tile i, then
// s_wait_asynccnt 0 + barrier closes the iteration.
// ---------------------------------------------------------------------------
__global__ __launch_bounds__(256) void proj_gemm_kernel(const _Float16* __restrict__ y16,
                                                        const _Float16* __restrict__ wp,
                                                        const float* __restrict__ pb,
                                                        float* __restrict__ out) {
    __shared__ __align__(16) _Float16 As[2][128 * 40];
    __shared__ __align__(16) _Float16 Bs[2][128 * 40];
    const int tid  = threadIdx.x;
    const int lane = tid & 31, wave = tid >> 5;
    const int l16  = lane & 15, hf = lane >> 4;
    const int wm = wave & 3;
    const int wn = wave >> 2;
    const int gm0 = blockIdx.y * 128;
    const int gn0 = blockIdx.x * 128;

    const f8 zf = {0.f,0.f,0.f,0.f,0.f,0.f,0.f,0.f};
    f8 acc[2][4];
#pragma unroll
    for (int mi = 0; mi < 2; ++mi)
#pragma unroll
        for (int nj = 0; nj < 4; ++nj) acc[mi][nj] = zf;

    // Prologue: async-load K tile 0 into buffer 0.
#pragma unroll
    for (int i = 0; i < 2; ++i) {
        int e = tid + i * 256, row = e >> 2, c = e & 3;
        async_cp16(&As[0][row * 40 + c * 8],
                   &y16[(size_t)(gm0 + row) * CDIM + c * 8]);
        async_cp16(&Bs[0][row * 40 + c * 8],
                   &wp[(size_t)(gn0 + row) * CDIM + c * 8]);
    }
    async_wait0();
    __syncthreads();

    for (int it = 0; it < CDIM / 32; ++it) {
        const int k0  = it * 32;
        const int cur = it & 1;
        // Issue DMA for the next tile into the alternate buffer.
        if (k0 + 32 < CDIM) {
#pragma unroll
            for (int i = 0; i < 2; ++i) {
                int e = tid + i * 256, row = e >> 2, c = e & 3;
                async_cp16(&As[cur ^ 1][row * 40 + c * 8],
                           &y16[(size_t)(gm0 + row) * CDIM + k0 + 32 + c * 8]);
                async_cp16(&Bs[cur ^ 1][row * 40 + c * 8],
                           &wp[(size_t)(gn0 + row) * CDIM + k0 + 32 + c * 8]);
            }
        }
        // Compute on current buffer while DMA fills the other one.
        h16 af[2], bf[4];
#pragma unroll
        for (int mi = 0; mi < 2; ++mi)
            af[mi] = fragA(As[cur], wm * 32 + mi * 16 + l16, 40, 0, hf);
#pragma unroll
        for (int nj = 0; nj < 4; ++nj)
            bf[nj] = fragB(Bs[cur], wn * 64 + nj * 16 + l16, 40, 0, hf);
#pragma unroll
        for (int mi = 0; mi < 2; ++mi)
#pragma unroll
            for (int nj = 0; nj < 4; ++nj)
                acc[mi][nj] = wmma_f16(af[mi], bf[nj], acc[mi][nj]);

        async_wait0();
        __syncthreads();
    }

    int   ncol[4];
    float bn[4];
#pragma unroll
    for (int nj = 0; nj < 4; ++nj) {
        ncol[nj] = gn0 + wn * 64 + nj * 16 + l16;
        bn[nj]   = pb[ncol[nj]];
    }
#pragma unroll
    for (int mi = 0; mi < 2; ++mi)
#pragma unroll
        for (int r = 0; r < 8; ++r) {
            int m = gm0 + wm * 32 + mi * 16 + r + hf * 8;
#pragma unroll
            for (int nj = 0; nj < 4; ++nj)
                out[(size_t)m * CDIM + ncol[nj]] = acc[mi][nj][r] + bn[nj];
        }
}

// ---------------------------------------------------------------------------
extern "C" void kernel_launch(void* const* d_in, const int* in_sizes, int n_in,
                              void* d_out, int out_size, void* d_ws, size_t ws_size,
                              hipStream_t stream) {
    const float* x      = (const float*)d_in[0];
    const float* mask   = (const float*)d_in[1];
    const float* qkv_w  = (const float*)d_in[2];
    const float* qkv_b  = (const float*)d_in[3];
    const float* proj_w = (const float*)d_in[4];
    const float* proj_b = (const float*)d_in[5];
    const float* rpb    = (const float*)d_in[6];

    const size_t P = (size_t)NWIN * NTOK * CDIM; // 102,760,448 elems per tensor
    _Float16* q16  = (_Float16*)d_ws;
    _Float16* k16  = q16 + P;
    _Float16* vt16 = k16 + P;
    _Float16* y16  = vt16 + P;
    _Float16* wq16 = y16 + P;
    _Float16* wp16 = wq16 + (size_t)3 * CDIM * CDIM;
    float*    biasf = (float*)(wp16 + (size_t)CDIM * CDIM);

    // 0) weight conversion + bias expand
    {
        int total = 3 * CDIM * CDIM + CDIM * CDIM + NH * NTOK * NTOK;
        prep_kernel<<<(total + 255) / 256, 256, 0, stream>>>(qkv_w, proj_w, rpb,
                                                             wq16, wp16, biasf);
    }
    // 1) QKV projection: (200704 x 512) @ (512 x 1536)
    qkv_gemm_kernel<<<dim3(12, 1568), 256, 0, stream>>>(x, wq16, qkv_b,
                                                        q16, k16, vt16);
    // 2) windowed attention
    attn_kernel<<<NWIN, 256, 0, stream>>>(q16, k16, vt16, mask, biasf, y16);
    // 3) output projection: (200704 x 512) @ (512 x 512)
    proj_gemm_kernel<<<dim3(4, 1568), 256, 0, stream>>>(y16, wp16, proj_b,
                                                        (float*)d_out);
}
